// FunctionalSegmentModel_37538014167559
// MI455X (gfx1250) — matrix operand order repaired and checked by reading, not compiled, not run on hardware
//
#include <hip/hip_runtime.h>
#include <hip/hip_bf16.h>
#include <math.h>

// Sizes from the reference
#define Bn 4
#define Tn 512
#define Fn 93
#define DSn 64
#define HSn 8
#define Dn 1024
#define Hn 8

typedef __attribute__((ext_vector_type(16))) _Float16 v16h;
typedef __attribute__((ext_vector_type(8)))  _Float16 v8h;
typedef __attribute__((ext_vector_type(8)))  float    v8f;

// Async copy: 16 bytes global -> LDS, tracked by ASYNCcnt (CDNA5 §10, opcode 98).
// VDST = per-lane LDS byte offset, VADDR = 64-bit global address.
__device__ __forceinline__ void async_ld_b128(unsigned int lds_off, const _Float16* g) {
  asm volatile("global_load_async_to_lds_b128 %0, %1, off"
               :: "v"(lds_off), "v"((unsigned long long)(uintptr_t)g)
               : "memory");
}

// ---------------------------------------------------------------------------
// K0: convert Wq/Wk/Wv/Wo (f32 [K][N]) -> f16 transposed [N][K] for WMMA B-frags
// ---------------------------------------------------------------------------
__global__ void conv_weights(const float* __restrict__ Wq, const float* __restrict__ Wk,
                             const float* __restrict__ Wv, const float* __restrict__ Wo,
                             _Float16* __restrict__ wT) {
  int idx = blockIdx.x * 256 + threadIdx.x;          // 4 * 1024 * 1024 total
  int wi  = idx >> 20;
  int r   = idx & 1048575;
  int n   = r >> 10;
  int k   = r & 1023;
  const float* src = (wi == 0) ? Wq : (wi == 1) ? Wk : (wi == 2) ? Wv : Wo;
  wT[((size_t)wi << 20) + ((size_t)n << 10) + k] = (_Float16)src[((size_t)k << 10) + n];
}

// ---------------------------------------------------------------------------
// K1: spectral stage, one block per (b,t). Only the fct row (i=0) of the
// spectral attention output survives into xt, and its mask row is all-true.
// ---------------------------------------------------------------------------
__global__ void spectral(const float* __restrict__ x, const float* __restrict__ xs,
                         const float* __restrict__ Wqs, const float* __restrict__ Wks,
                         const float* __restrict__ Wvs, const float* __restrict__ Wos,
                         const float* __restrict__ Wt, _Float16* __restrict__ xt) {
  __shared__ __align__(32) _Float16 sX[96][64];       // A matrix (rows 93..95 zero)
  __shared__ __align__(32) _Float16 sWT[2][64][64];   // [0]=Wks^T, [1]=Wvs^T  ([n][k])
  __shared__ __align__(32) _Float16 sK[96][64];
  __shared__ __align__(32) _Float16 sV[96][64];
  __shared__ float sq0[64];
  __shared__ float sSc[8][96];
  __shared__ float sInv[8];
  __shared__ float sY0[64];
  __shared__ float sOut[64];

  const int tid  = threadIdx.x;
  const int lane = tid & 31;
  const int wv   = tid >> 5;
  const int bt   = blockIdx.x;                        // b*T + t
  const size_t xsBase = (size_t)bt * (Fn * DSn);
  const size_t xBase  = (size_t)bt * Dn;

  for (int i = tid; i < 96 * 64; i += 256) {
    int r = i >> 6, k = i & 63;
    float v = (r < Fn) ? xs[xsBase + (size_t)r * 64 + k] : 0.0f;
    sX[r][k] = (_Float16)v;
  }
  for (int i = tid; i < 2 * 64 * 64; i += 256) {
    int g = i >> 12, r = i & 4095;
    int k = r >> 6, n = r & 63;
    const float* W = g ? Wvs : Wks;
    sWT[g][n][k] = (_Float16)W[(size_t)k * 64 + n];
  }
  __syncthreads();

  // ---- WMMA: K_s and V_s ----
  {
    const int m    = lane & 15;
    const int half = lane >> 4;
    for (int wi = wv; wi < 48; wi += 8) {
      const int g  = wi / 24;
      const int t  = wi % 24;
      const int mt = t >> 2, nt = t & 3;
      v8f acc = {};
      #pragma unroll
      for (int k0 = 0; k0 < 64; k0 += 32) {
        v8h alo = *(const v8h*)&sX[mt * 16 + m][k0 + half * 16];
        v8h ahi = *(const v8h*)&sX[mt * 16 + m][k0 + half * 16 + 8];
        v16h a = __builtin_shufflevector(alo, ahi, 0, 1, 2, 3, 4, 5, 6, 7,
                                                   8, 9, 10, 11, 12, 13, 14, 15);
        v16h bf = *(const v16h*)&sWT[g][nt * 16 + m][k0 + half * 16];
        acc = __builtin_amdgcn_wmma_f32_16x16x32_f16(false, a, false, bf,
                                                     (short)0, acc, false, false);
      }
      _Float16* dst = g ? &sV[0][0] : &sK[0][0];
      #pragma unroll
      for (int r = 0; r < 8; ++r)
        dst[(mt * 16 + half * 8 + r) * 64 + nt * 16 + m] = (_Float16)acc[r];
    }
  }
  // ---- q0 = X[0] @ Wqs ----
  if (tid < 64) {
    float acc = 0.0f;
    #pragma unroll 8
    for (int k = 0; k < 64; ++k) acc += (float)sX[0][k] * Wqs[(size_t)k * 64 + tid];
    sq0[tid] = acc;
  }
  __syncthreads();

  for (int i = tid; i < HSn * Fn; i += 256) {
    int h = i / Fn, j = i % Fn;
    float s = 0.0f;
    #pragma unroll
    for (int d = 0; d < 8; ++d) s += sq0[h * 8 + d] * (float)sK[j][h * 8 + d];
    sSc[h][j] = s * 0.35355339059327373f;
  }
  __syncthreads();

  if (tid < 8) {
    float m = -1e30f;
    for (int j = 0; j < Fn; ++j) m = fmaxf(m, sSc[tid][j]);
    float sum = 0.0f;
    for (int j = 0; j < Fn; ++j) { float e = expf(sSc[tid][j] - m); sSc[tid][j] = e; sum += e; }
    sInv[tid] = 1.0f / sum;
  }
  __syncthreads();

  if (tid < 64) {
    int h = tid >> 3, d = tid & 7;
    float acc = 0.0f;
    for (int j = 0; j < Fn; ++j) acc += sSc[h][j] * (float)sV[j][h * 8 + d];
    sY0[tid] = acc * sInv[h];
  }
  __syncthreads();

  if (tid < 64) {
    float acc = 0.0f;
    #pragma unroll 8
    for (int k = 0; k < 64; ++k) acc += sY0[k] * Wos[(size_t)k * 64 + tid];
    sOut[tid] = acc;
  }
  __syncthreads();

  for (int c = tid; c < Dn; c += 256) {
    float acc = x[xBase + c];
    #pragma unroll 8
    for (int k = 0; k < 64; ++k) acc += sOut[k] * Wt[(size_t)k * Dn + c];
    xt[xBase + c] = (_Float16)acc;
  }
}

// ---------------------------------------------------------------------------
// K2/K4: GEMM  C[2048,1024] = A[2048,1024](f16) x W^T[N][K](f16).
// 128x128 tile per block; wave owns a 16x128 strip (8 WMMA accumulators).
// Double-buffered LDS tiles filled with GLOBAL_LOAD_ASYNC_TO_LDS_B128
// (ASYNCcnt), so the next k-tile streams in while WMMAs run.
// Dynamic LDS layout (32 KB): buf p at p*16384: sA[128][32] then sB[128][32].
// ---------------------------------------------------------------------------
__global__ void gemm128(const _Float16* __restrict__ A, const _Float16* __restrict__ WTbase,
                        _Float16* __restrict__ outHbase, float* __restrict__ outF) {
  extern __shared__ __align__(32) char smem[];

  const int tid  = threadIdx.x;
  const int lane = tid & 31;
  const int wv   = tid >> 5;
  const int m    = lane & 15;
  const int half = lane >> 4;
  const size_t z = blockIdx.z;
  const _Float16* WT = WTbase + (z << 20);
  _Float16* outH = outHbase ? outHbase + z * (size_t)(2048 * 1024) : (_Float16*)0;

  const size_t rowBase = (size_t)blockIdx.y * 128;
  const size_t colBase = (size_t)blockIdx.x * 128;

  // per-thread tile-load geometry: 2 chunks of 8 halves for A, same for B
  const int ch0 = tid * 2;
  const int r0  = ch0 >> 2;
  const int kk0 = (ch0 & 3) * 8;
  const int r1  = (ch0 + 1) >> 2;
  const int kk1 = ((ch0 + 1) & 3) * 8;

  v8f acc[8];
  #pragma unroll
  for (int i = 0; i < 8; ++i) acc[i] = (v8f){};

  // prologue: issue tile 0 into buffer 0
  {
    unsigned int pb = 0;
    async_ld_b128(pb + r0 * 64 + kk0 * 2,        A  + (rowBase + r0) * 1024 + kk0);
    async_ld_b128(pb + 8192 + r0 * 64 + kk0 * 2, WT + (colBase + r0) * 1024 + kk0);
    async_ld_b128(pb + r1 * 64 + kk1 * 2,        A  + (rowBase + r1) * 1024 + kk1);
    async_ld_b128(pb + 8192 + r1 * 64 + kk1 * 2, WT + (colBase + r1) * 1024 + kk1);
  }

  for (int it = 0; it < 32; ++it) {
    const int p = it & 1;
    const unsigned int pb = (unsigned int)p * 16384u;

    if (it < 31) {
      const int k0n = (it + 1) * 32;
      const unsigned int nb = (unsigned int)(1 - p) * 16384u;
      async_ld_b128(nb + r0 * 64 + kk0 * 2,        A  + (rowBase + r0) * 1024 + k0n + kk0);
      async_ld_b128(nb + 8192 + r0 * 64 + kk0 * 2, WT + (colBase + r0) * 1024 + k0n + kk0);
      async_ld_b128(nb + r1 * 64 + kk1 * 2,        A  + (rowBase + r1) * 1024 + k0n + kk1);
      async_ld_b128(nb + 8192 + r1 * 64 + kk1 * 2, WT + (colBase + r1) * 1024 + k0n + kk1);
      asm volatile("s_wait_asynccnt 0x4" ::: "memory");  // current tile's 4 ops done
    } else {
      asm volatile("s_wait_asynccnt 0x0" ::: "memory");
    }
    __syncthreads();   // all waves' async writes for tile `it` visible

    const _Float16* sA = (const _Float16*)(smem + pb);
    const _Float16* sB = (const _Float16*)(smem + pb + 8192);

    v8h alo = *(const v8h*)&sA[((wv << 4) + m) * 32 + half * 16];
    v8h ahi = *(const v8h*)&sA[((wv << 4) + m) * 32 + half * 16 + 8];
    v16h a = __builtin_shufflevector(alo, ahi, 0, 1, 2, 3, 4, 5, 6, 7,
                                               8, 9, 10, 11, 12, 13, 14, 15);
    #pragma unroll
    for (int nt = 0; nt < 8; ++nt) {
      v16h bf = *(const v16h*)&sB[(nt * 16 + m) * 32 + half * 16];
      acc[nt] = __builtin_amdgcn_wmma_f32_16x16x32_f16(false, a, false, bf,
                                                       (short)0, acc[nt], false, false);
    }
    __syncthreads();   // buffer `p` free for reuse two iterations later
  }

  #pragma unroll
  for (int nt = 0; nt < 8; ++nt) {
    #pragma unroll
    for (int r = 0; r < 8; ++r) {
      size_t row = rowBase + (wv << 4) + (half << 3) + r;
      size_t col = colBase + nt * 16 + m;
      size_t off = row * 1024 + col;
      float v = acc[nt][r];
      if (outF) outF[off] = v;
      else      outH[off] = (_Float16)v;
    }
  }
}

// ---------------------------------------------------------------------------
// K3: temporal attention, one wave per query (b,h,i). The per-head partition
// mask is a contiguous j-range with closed-form bounds; pads attend [l,T).
// ---------------------------------------------------------------------------
__global__ void temporal_attn(const _Float16* __restrict__ q, const _Float16* __restrict__ k,
                              const _Float16* __restrict__ v, const int* __restrict__ valid_len,
                              _Float16* __restrict__ attnOut) {
  __shared__ float sS[8][512];
  __shared__ float sQ[8][128];

  const int lane = threadIdx.x & 31;
  const int wv   = threadIdx.x >> 5;
  const int g    = blockIdx.x * 8 + wv;              // 0 .. B*H*T-1
  const int b    = g >> 12;                          // H*T = 4096
  const int rem  = g & 4095;
  const int h    = rem >> 9;
  const int i    = rem & 511;

  const int l   = valid_len[b];
  const int dv  = (h >> 1) + 1;
  int js, je;
  if (i < l) {
    int c = (i * dv) / l;
    js = (c * l + dv - 1) / dv;
    int e = ((c + 1) * l + dv - 1) / dv;
    je = (e < l) ? e : l;
  } else {
    js = l; je = Tn;
  }
  const int n = je - js;

  const size_t qoff = ((size_t)(b * Tn + i)) * Dn + (size_t)h * 128;
  for (int d = lane; d < 128; d += 32) sQ[wv][d] = (float)q[qoff + d];
  __syncthreads();

  // pass 1: scores + running max (prefetch matching V rows for pass 2)
  float mloc = -1e30f;
  for (int jj = lane; jj < n; jj += 32) {
    int j = js + jj;
    const size_t roff = ((size_t)(b * Tn + j)) * Dn + (size_t)h * 128;
    __builtin_prefetch(v + roff, 0, 1);              // global_prefetch_b8
    const v8h* kr = (const v8h*)(k + roff);
    float s = 0.0f;
    #pragma unroll
    for (int c = 0; c < 16; ++c) {
      v8h kv = kr[c];
      #pragma unroll
      for (int u = 0; u < 8; ++u) s += sQ[wv][c * 8 + u] * (float)kv[u];
    }
    s *= 0.08838834764831845f;                       // 1/sqrt(128)
    sS[wv][jj] = s;
    mloc = fmaxf(mloc, s);
  }
  #pragma unroll
  for (int o = 16; o > 0; o >>= 1) mloc = fmaxf(mloc, __shfl_xor(mloc, o, 32));

  float sloc = 0.0f;
  for (int jj = lane; jj < n; jj += 32) {
    float e = expf(sS[wv][jj] - mloc);
    sS[wv][jj] = e;
    sloc += e;
  }
  #pragma unroll
  for (int o = 16; o > 0; o >>= 1) sloc += __shfl_xor(sloc, o, 32);
  const float inv = 1.0f / sloc;
  __syncthreads();

  float acc0 = 0.f, acc1 = 0.f, acc2 = 0.f, acc3 = 0.f;
  for (int jj = 0; jj < n; ++jj) {
    float p = sS[wv][jj];
    const _Float16* vr = v + ((size_t)(b * Tn + js + jj)) * Dn + (size_t)h * 128;
    acc0 += p * (float)vr[lane];
    acc1 += p * (float)vr[lane + 32];
    acc2 += p * (float)vr[lane + 64];
    acc3 += p * (float)vr[lane + 96];
  }
  _Float16* orow = attnOut + qoff;
  orow[lane]      = (_Float16)(acc0 * inv);
  orow[lane + 32] = (_Float16)(acc1 * inv);
  orow[lane + 64] = (_Float16)(acc2 * inv);
  orow[lane + 96] = (_Float16)(acc3 * inv);
}

// ---------------------------------------------------------------------------
// Host side
// ---------------------------------------------------------------------------
extern "C" void kernel_launch(void* const* d_in, const int* in_sizes, int n_in,
                              void* d_out, int out_size, void* d_ws, size_t ws_size,
                              hipStream_t stream) {
  const float* x   = (const float*)d_in[0];
  const float* xs  = (const float*)d_in[1];
  const int*   vl  = (const int*)d_in[2];
  const float* Wq  = (const float*)d_in[3];
  const float* Wk  = (const float*)d_in[4];
  const float* Wv  = (const float*)d_in[5];
  const float* Wo  = (const float*)d_in[6];
  const float* Wqs = (const float*)d_in[7];
  const float* Wks = (const float*)d_in[8];
  const float* Wvs = (const float*)d_in[9];
  const float* Wos = (const float*)d_in[10];
  const float* Wt  = (const float*)d_in[11];
  float* out = (float*)d_out;

  char* ws = (char*)d_ws;
  const size_t MB = 1024 * 1024;
  _Float16* xt  = (_Float16*)(ws + 0);          //  4 MB : xt  f16 [2048][1024]
  _Float16* qb  = (_Float16*)(ws + 4  * MB);    //  4 MB : Q   f16
  _Float16* kb  = (_Float16*)(ws + 8  * MB);    //  4 MB : K   f16
  _Float16* vb  = (_Float16*)(ws + 12 * MB);    //  4 MB : V   f16
  _Float16* wT  = (_Float16*)(ws + 16 * MB);    //  8 MB : WqT/WkT/WvT/WoT f16
  _Float16* att = (_Float16*)(ws + 24 * MB);    //  4 MB : attention output f16

  conv_weights<<<16384, 256, 0, stream>>>(Wq, Wk, Wv, Wo, wT);
  spectral<<<Bn * Tn, 256, 0, stream>>>(x, xs, Wqs, Wks, Wvs, Wos, Wt, xt);
  gemm128<<<dim3(8, 16, 3), 256, 32768, stream>>>(xt, wT, qb, nullptr);
  temporal_attn<<<Bn * Hn * Tn / 8, 256, 0, stream>>>(qb, kb, vb, vl, att);
  gemm128<<<dim3(8, 16, 1), 256, 32768, stream>>>(att, wT + 3 * (size_t)(1024 * 1024), nullptr, out);
}
